// Attention_82429012345336
// MI455X (gfx1250) — matrix operand order repaired and checked by reading, not compiled, not run on hardware
//
#include <hip/hip_runtime.h>
#include <hip/hip_bf16.h>

typedef __attribute__((ext_vector_type(16))) __bf16 bf16x16;
typedef __attribute__((ext_vector_type(8)))  __bf16 bf16x8;
typedef __attribute__((ext_vector_type(2)))  __bf16 bf16x2v;
typedef __attribute__((ext_vector_type(2)))  float  f32x2;
typedef __attribute__((ext_vector_type(8)))  float  f32x8;

#define BM 128
#define BKK 32
#define LPAD 8
#define THREADS 256

// pack two floats into two bf16 (low = a, high = b) -> v_cvt_pk_bf16_f32
__device__ __forceinline__ unsigned pack2bf(float a, float b) {
#if __has_builtin(__builtin_amdgcn_cvt_pk_bf16_f32)
  return __builtin_bit_cast(unsigned, __builtin_amdgcn_cvt_pk_bf16_f32(a, b));
#else
  f32x2 f = {a, b};
  bf16x2v r = __builtin_convertvector(f, bf16x2v);  // fptrunc v2f32 -> v2bf16
  return __builtin_bit_cast(unsigned, r);
#endif
}

// C = (ascale*A) x B (bf16 WMMA, fp32 accumulate). All dims exact multiples of
// tile (BM=128 in M, BNT in N, BKK=32 in K) -- no bounds checks.
// A element (m,k) = ATR ? A[k*lda+m] : A[m*lda+k]
// B element (k,n) = BTR ? B[n*ldb+k] : B[k*ldb+n]
// Batch: z = blockIdx.z; outer = z/innerMod, inner = z%innerMod.
// Double-buffered LDS, one barrier per K-step, last iteration peeled,
// staging split into load-phase / convert-store-phase so all global_load_b128
// are in flight together.
template<bool ATR, bool BTR, int BNT>
__global__ __launch_bounds__(THREADS)
void wmma_gemm(const float* __restrict__ A, const float* __restrict__ Bm,
               float* __restrict__ C, float ascale,
               int Kr, int lda, int ldb, int ldc, int innerMod,
               long long sAo, long long sAi,
               long long sBo, long long sBi,
               long long sCo, long long sCi)
{
  constexpr int NI = BNT / 32;      // wmma n-tiles per wave (wave covers BNT/2)
  constexpr int AIT = (BM*BKK)/(THREADS*4);   // 4
  constexpr int BIT = (BNT*BKK)/(THREADS*4);  // BNT/32
  __shared__ __align__(16) unsigned short As[2][BM][BKK + LPAD];
  __shared__ __align__(16) unsigned short Bs[2][BNT][BKK + LPAD];

  const int z = blockIdx.z;
  const int outer = z / innerMod, inner = z % innerMod;
  const float* Ab = A  + (long long)outer * sAo + (long long)inner * sAi;
  const float* Bb = Bm + (long long)outer * sBo + (long long)inner * sBi;
  float*       Cb = C  + (long long)outer * sCo + (long long)inner * sCi;

  const int m0 = blockIdx.y * BM;
  const int n0 = blockIdx.x * BNT;
  const int tid  = threadIdx.x;
  const int wid  = tid >> 5;
  const int lane = tid & 31;
  const int half = lane >> 4;
  const int lrow = lane & 15;
  const int wm = (wid >> 1) * 32;        // wave origin in M: 0,32,64,96
  const int wn = (wid & 1) * (BNT / 2);  // wave origin in N

  auto stage = [&](int k0, int buf) {
    float4 fa[AIT];
    float4 fb[BIT];
    // ---- load phase: issue every global_load_b128 before any wait ----
    if constexpr (!ATR) {
#pragma unroll
      for (int i = 0; i < AIT; ++i) {
        int e = tid + i*THREADS;
        int r = e >> 3, c4 = (e & 7) << 2;
        fa[i] = *(const float4*)(Ab + (size_t)(m0 + r)*lda + (k0 + c4));
      }
    } else {
#pragma unroll
      for (int i = 0; i < AIT; ++i) {
        int e = tid + i*THREADS;
        int m4 = (e & 31) << 2, kk = e >> 5;
        fa[i] = *(const float4*)(Ab + (size_t)(k0 + kk)*lda + (m0 + m4));
      }
    }
    if constexpr (BTR) {
#pragma unroll
      for (int i = 0; i < BIT; ++i) {
        int e = tid + i*THREADS;
        int n = e >> 3, c4 = (e & 7) << 2;
        fb[i] = *(const float4*)(Bb + (size_t)(n0 + n)*ldb + (k0 + c4));
      }
    } else {
#pragma unroll
      for (int i = 0; i < BIT; ++i) {
        int e = tid + i*THREADS;
        int n = e & (BNT-1), kk4 = (e / BNT) << 2;
        const float* bp = Bb + (size_t)(k0 + kk4)*ldb + (n0 + n);
        fb[i].x = bp[0];
        fb[i].y = bp[(size_t)ldb];
        fb[i].z = bp[2*(size_t)ldb];
        fb[i].w = bp[3*(size_t)ldb];
      }
    }
    // prefetch next tile -> global_prefetch_b8
    if (k0 + BKK < Kr) {
      if constexpr (!ATR)
        __builtin_prefetch(Ab + (size_t)(m0 + (tid >> 1))*lda + k0 + BKK, 0, 1);
      else
        __builtin_prefetch(Ab + (size_t)(k0 + BKK + (tid & 31))*lda + m0, 0, 1);
      if constexpr (BTR)
        __builtin_prefetch(Bb + (size_t)(n0 + (tid & (BNT-1)))*ldb + k0 + BKK, 0, 1);
      else
        __builtin_prefetch(Bb + (size_t)(k0 + BKK + (tid & 31))*ldb + n0, 0, 1);
    }
    // ---- convert + LDS store phase ----
    if constexpr (!ATR) {
#pragma unroll
      for (int i = 0; i < AIT; ++i) {
        int e = tid + i*THREADS;
        int r = e >> 3, c4 = (e & 7) << 2;
        *(uint2*)&As[buf][r][c4] =
            make_uint2(pack2bf(fa[i].x*ascale, fa[i].y*ascale),
                       pack2bf(fa[i].z*ascale, fa[i].w*ascale));
      }
    } else {
#pragma unroll
      for (int i = 0; i < AIT; ++i) {
        int e = tid + i*THREADS;
        int m4 = (e & 31) << 2, kk = e >> 5;
        unsigned p01 = pack2bf(fa[i].x*ascale, fa[i].y*ascale);
        unsigned p23 = pack2bf(fa[i].z*ascale, fa[i].w*ascale);
        As[buf][m4+0][kk] = (unsigned short)p01;
        As[buf][m4+1][kk] = (unsigned short)(p01 >> 16);
        As[buf][m4+2][kk] = (unsigned short)p23;
        As[buf][m4+3][kk] = (unsigned short)(p23 >> 16);
      }
    }
    if constexpr (BTR) {
#pragma unroll
      for (int i = 0; i < BIT; ++i) {
        int e = tid + i*THREADS;
        int n = e >> 3, c4 = (e & 7) << 2;
        *(uint2*)&Bs[buf][n][c4] =
            make_uint2(pack2bf(fb[i].x, fb[i].y), pack2bf(fb[i].z, fb[i].w));
      }
    } else {
#pragma unroll
      for (int i = 0; i < BIT; ++i) {
        int e = tid + i*THREADS;
        int n = e & (BNT-1), kk4 = (e / BNT) << 2;
        *(uint2*)&Bs[buf][n][kk4] =
            make_uint2(pack2bf(fb[i].x, fb[i].y), pack2bf(fb[i].z, fb[i].w));
      }
    }
  };

  f32x8 acc[2][NI];
#pragma unroll
  for (int mi = 0; mi < 2; ++mi)
#pragma unroll
    for (int ni = 0; ni < NI; ++ni)
      acc[mi][ni] = (f32x8){0.f,0.f,0.f,0.f,0.f,0.f,0.f,0.f};

  auto compute = [&](int cur) {
    // ---- fragments (ISA 7.12.2 16-bit layouts) ----
    union Frag { bf16x16 v; bf16x8 h[2]; } af[2], bfg[NI];
#pragma unroll
    for (int mi = 0; mi < 2; ++mi) {
      const unsigned short* p = &As[cur][wm + mi*16 + lrow][half*8];
      af[mi].h[0] = *(const bf16x8*)(p);        // K = koff..koff+7
      af[mi].h[1] = *(const bf16x8*)(p + 16);   // K = koff+16..koff+23
    }
#pragma unroll
    for (int ni = 0; ni < NI; ++ni) {
      const unsigned short* p = &Bs[cur][wn + ni*16 + lrow][half*8];
      bfg[ni].h[0] = *(const bf16x8*)(p);
      bfg[ni].h[1] = *(const bf16x8*)(p + 16);
    }
#pragma unroll
    for (int mi = 0; mi < 2; ++mi)
#pragma unroll
      for (int ni = 0; ni < NI; ++ni)
        acc[mi][ni] = __builtin_amdgcn_wmma_f32_16x16x32_bf16(
            false, af[mi].v, false, bfg[ni].v,
            (short)0, acc[mi][ni], false, false);
  };

  const int ktiles = Kr / BKK;
  stage(0, 0);
  for (int kt = 0; kt < ktiles - 1; ++kt) {     // steady state: straight-line
    __syncthreads();                            // buf[kt&1] ready
    stage((kt + 1) * BKK, (kt + 1) & 1);
    compute(kt & 1);
  }
  __syncthreads();
  compute((ktiles - 1) & 1);                    // peeled last iteration

  // ---- epilogue: C/D layout VGPR r -> M = half*8 + r, N = lrow ----
#pragma unroll
  for (int mi = 0; mi < 2; ++mi)
#pragma unroll
    for (int ni = 0; ni < NI; ++ni) {
      int col = n0 + wn + ni*16 + lrow;
#pragma unroll
      for (int r = 0; r < 8; ++r) {
        int row = m0 + wm + mi*16 + half*8 + r;
        Cb[(size_t)row*ldc + col] = acc[mi][ni][r];
      }
    }
}

// ---- rotary embedding, in place on q and k ----
__global__ __launch_bounds__(256)
void rotary_kernel(float* __restrict__ q, float* __restrict__ k,
                   const float* __restrict__ sinu,
                   int total_pairs, int L_, int H_, int Dh_)
{
  int idx = blockIdx.x*blockDim.x + threadIdx.x;
  if (idx >= total_pairs) return;
  int jp = Dh_ >> 1;
  int j = idx % jp;
  int t = idx / jp;                 // (b*L + l)*H + h
  int l = (t / H_) % L_;
  float s = sinu[l*Dh_ + j];
  float c = sinu[l*Dh_ + jp + j];
  size_t base = (size_t)t * Dh_ + 2*j;
  float a = q[base], b = q[base+1];
  q[base]   = a*c - b*s;
  q[base+1] = b*c + a*s;
  a = k[base]; b = k[base+1];
  k[base]   = a*c - b*s;
  k[base+1] = b*c + a*s;
}

__device__ __forceinline__ float wave_max(float v) {
#pragma unroll
  for (int off = 16; off > 0; off >>= 1) v = fmaxf(v, __shfl_xor(v, off, 32));
  return v;
}
__device__ __forceinline__ float wave_sum(float v) {
#pragma unroll
  for (int off = 16; off > 0; off >>= 1) v += __shfl_xor(v, off, 32);
  return v;
}

// dd is already scaled by data_normalizer (folded into GEMM A staging).
// qp[r,:] = ratio*(exp(dd - diag - rowmax) + eps), one wave per row (M=256)
__global__ __launch_bounds__(256)
void finalize_qp_kernel(float* __restrict__ qp, const float* __restrict__ q,
                        int rows, float normzr, float ratio)
{
  int wid = blockIdx.x * 8 + (threadIdx.x >> 5);
  int lane = threadIdx.x & 31;
  if (wid >= rows) return;
  size_t base = (size_t)wid * 256;
  float vals[8], mx = -3.402823466e38f;
#pragma unroll
  for (int j = 0; j < 8; ++j) {
    vals[j] = qp[base + lane + 32*j];
    mx = fmaxf(mx, vals[j]);
  }
  mx = wave_max(mx);
  size_t qb = (size_t)wid * 64;
  float a = q[qb + lane], b = q[qb + lane + 32];
  float ss = wave_sum(a*a + b*b);
  float diag = ss * 0.5f * normzr * normzr;
#pragma unroll
  for (int j = 0; j < 8; ++j)
    qp[base + lane + 32*j] = ratio * (__expf(vals[j] - diag - mx) + 1e-6f);
}

// per-row max of dd_k (already normalized)
__global__ __launch_bounds__(256)
void rowmax_kernel(const float* __restrict__ kp, float* __restrict__ rowmax,
                   int rows)
{
  int wid = blockIdx.x * 8 + (threadIdx.x >> 5);
  int lane = threadIdx.x & 31;
  if (wid >= rows) return;
  size_t base = (size_t)wid * 256;
  float mx = -3.402823466e38f;
#pragma unroll
  for (int j = 0; j < 8; ++j) mx = fmaxf(mx, kp[base + lane + 32*j]);
  mx = wave_max(mx);
  if (lane == 0) rowmax[wid] = mx;
}

// max over L per (b,h)
__global__ __launch_bounds__(256)
void bhmax_kernel(const float* __restrict__ rowmax, float* __restrict__ bhmax,
                  int L_, int H_)
{
  __shared__ float sm[256];
  int zidx = blockIdx.x;
  int b = zidx / H_, h = zidx % H_;
  float mx = -3.402823466e38f;
  for (int l = threadIdx.x; l < L_; l += 256)
    mx = fmaxf(mx, rowmax[((size_t)(b*L_ + l))*H_ + h]);
  sm[threadIdx.x] = mx;
  __syncthreads();
  for (int s = 128; s > 0; s >>= 1) {
    if ((int)threadIdx.x < s) sm[threadIdx.x] = fmaxf(sm[threadIdx.x], sm[threadIdx.x + s]);
    __syncthreads();
  }
  if (threadIdx.x == 0) bhmax[zidx] = sm[0];
}

__global__ __launch_bounds__(256)
void finalize_kp_kernel(float* __restrict__ kp, const float* __restrict__ k,
                        const float* __restrict__ bhmax,
                        int rows, int L_, int H_, float normzr, float ratio)
{
  int wid = blockIdx.x * 8 + (threadIdx.x >> 5);
  int lane = threadIdx.x & 31;
  if (wid >= rows) return;
  int h = wid % H_;
  int b = wid / (L_ * H_);
  float mx = bhmax[b*H_ + h];
  size_t base = (size_t)wid * 256;
  size_t kb = (size_t)wid * 64;
  float a = k[kb + lane], bb = k[kb + lane + 32];
  float ss = wave_sum(a*a + bb*bb);
  float diag = ss * 0.5f * normzr * normzr;
#pragma unroll
  for (int j = 0; j < 8; ++j) {
    float v = kp[base + lane + 32*j];
    kp[base + lane + 32*j] = ratio * (__expf(v - diag - mx) + 1e-6f);
  }
}

// ks_sum[b,h,m] = sum_l kp[b,l,h,m]
__global__ __launch_bounds__(256)
void kssum_kernel(const float* __restrict__ kp, float* __restrict__ kssum,
                  int B_, int L_, int H_, int M_)
{
  int idx = blockIdx.x*blockDim.x + threadIdx.x;
  if (idx >= B_*H_*M_) return;
  int m = idx % M_;
  int h = (idx / M_) % H_;
  int b = idx / (M_*H_);
  float s = 0.f;
  for (int l = 0; l < L_; ++l)
    s += kp[(((size_t)(b*L_ + l))*H_ + h)*M_ + m];
  kssum[idx] = s;
}

// av[r,:] /= (qp[r,:] . ks_sum[b,h,:]) , one wave per row
__global__ __launch_bounds__(256)
void divide_kernel(float* __restrict__ av, const float* __restrict__ qp,
                   const float* __restrict__ kssum,
                   int rows, int L_, int H_, int M_, int Dh_)
{
  int wid = blockIdx.x * 8 + (threadIdx.x >> 5);
  int lane = threadIdx.x & 31;
  if (wid >= rows) return;
  int h = wid % H_;
  int b = wid / (L_ * H_);
  size_t qb = (size_t)wid * M_;
  const float* ks = kssum + ((size_t)(b*H_ + h)) * M_;
  float s = 0.f;
#pragma unroll
  for (int j = 0; j < 8; ++j)
    s += qp[qb + lane + 32*j] * ks[lane + 32*j];
  s = wave_sum(s);
  float inv = 1.0f / s;
  size_t ab = (size_t)wid * Dh_;
  av[ab + lane]      *= inv;
  av[ab + lane + 32] *= inv;
}

extern "C" void kernel_launch(void* const* d_in, const int* in_sizes, int n_in,
                              void* d_out, int out_size, void* d_ws, size_t ws_size,
                              hipStream_t stream) {
  (void)in_sizes; (void)n_in; (void)out_size; (void)ws_size;
  const float* x    = (const float*)d_in[0];
  const float* Wq   = (const float*)d_in[1];
  const float* Wk   = (const float*)d_in[2];
  const float* Wv   = (const float*)d_in[3];
  const float* Wo   = (const float*)d_in[4];
  const float* proj = (const float*)d_in[5];
  const float* sinu = (const float*)d_in[6];
  float* out = (float*)d_out;

  constexpr int Bb = 2, Ls = 4096, Dm = 1024, Hh = 16, Dhv = 64, Mm = 256;
  const float normzr = 0.35355339059327373f;  // 64^-0.25
  const float ratio  = 0.0625f;               // 256^-0.5

  const size_t nQ = (size_t)Bb*Ls*Dm;        // 8,388,608
  const size_t nP = (size_t)Bb*Ls*Hh*Mm;     // 33,554,432
  float* ws = (float*)d_ws;
  float* q      = ws;
  float* k      = q  + nQ;
  float* v      = k  + nQ;
  float* qp     = v  + nQ;
  float* kp     = qp + nP;
  float* rowmax = kp + nP;                    // B*L*H
  float* bhmax  = rowmax + (size_t)Bb*Ls*Hh;  // B*H
  float* kssum  = bhmax  + (size_t)Bb*Hh;     // B*H*M
  float* kvs    = kssum  + (size_t)Bb*Hh*Mm;  // B*H*M*Dh
  float* av     = kvs    + (size_t)Bb*Hh*Mm*Dhv;

  dim3 blk(THREADS);
  const int rows = Bb*Ls*Hh;                  // 131072

  // 1-3) q,k,v = x @ W{q,k,v}   (8192 x 1024 x 1024)
  dim3 g1(Dm/128, (Bb*Ls)/BM, 1);
  wmma_gemm<false,false,128><<<g1, blk, 0, stream>>>(x, Wq, q, 1.0f, Dm, Dm, Dm, Dm,
                                                     1, 0,0, 0,0, 0,0);
  wmma_gemm<false,false,128><<<g1, blk, 0, stream>>>(x, Wk, k, 1.0f, Dm, Dm, Dm, Dm,
                                                     1, 0,0, 0,0, 0,0);
  wmma_gemm<false,false,128><<<g1, blk, 0, stream>>>(x, Wv, v, 1.0f, Dm, Dm, Dm, Dm,
                                                     1, 0,0, 0,0, 0,0);

  // 4) rotary on q,k
  int pairs = Bb*Ls*Hh*(Dhv/2);
  rotary_kernel<<<pairs/256, blk, 0, stream>>>(q, k, sinu, pairs, Ls, Hh, Dhv);

  // 5-6) dd = (normzr*(q|k)) @ proj^T  (131072 x 256 x 64), proj (M,Dh) -> BTR
  dim3 g2(Mm/128, rows/BM, 1);
  wmma_gemm<false,true,128><<<g2, blk, 0, stream>>>(q, proj, qp, normzr,
                                                    Dhv, Dhv, Dhv, Mm,
                                                    1, 0,0, 0,0, 0,0);
  wmma_gemm<false,true,128><<<g2, blk, 0, stream>>>(k, proj, kp, normzr,
                                                    Dhv, Dhv, Dhv, Mm,
                                                    1, 0,0, 0,0, 0,0);

  // 7) query softmax-kernel finalize (per-row max)
  finalize_qp_kernel<<<rows/8, blk, 0, stream>>>(qp, q, rows, normzr, ratio);

  // 8-10) key softmax-kernel finalize (global per-(b,h) max)
  rowmax_kernel<<<rows/8, blk, 0, stream>>>(kp, rowmax, rows);
  bhmax_kernel<<<Bb*Hh, blk, 0, stream>>>(rowmax, bhmax, Ls, Hh);
  finalize_kp_kernel<<<rows/8, blk, 0, stream>>>(kp, k, bhmax, rows, Ls, Hh, normzr, ratio);

  // 11) ks_sum[b,h,m]
  kssum_kernel<<<(Bb*Hh*Mm)/256, blk, 0, stream>>>(kp, kssum, Bb, Ls, Hh, Mm);

  // 12) kvs[b,h] = kp[b,:,h,:]^T @ v[b,:,h,:]   (256 x 64 x 4096) per (b,h)
  dim3 g3(Dhv/64, Mm/BM, Bb*Hh);
  wmma_gemm<true,false,64><<<g3, blk, 0, stream>>>(kp, v, kvs, 1.0f,
      Ls, Hh*Mm, Hh*Dhv, Dhv, Hh,
      (long long)Ls*Hh*Mm, (long long)Mm,
      (long long)Ls*Hh*Dhv, (long long)Dhv,
      (long long)Hh*Mm*Dhv, (long long)Mm*Dhv);

  // 13) av[b,:,h,:] = qp[b,:,h,:] @ kvs[b,h]   (4096 x 64 x 256) per (b,h)
  dim3 g4(Dhv/64, Ls/BM, Bb*Hh);
  wmma_gemm<false,false,64><<<g4, blk, 0, stream>>>(qp, kvs, av, 1.0f,
      Mm, Hh*Mm, Dhv, Hh*Dhv, Hh,
      (long long)Ls*Hh*Mm, (long long)Mm,
      (long long)Hh*Mm*Dhv, (long long)Mm*Dhv,
      (long long)Ls*Hh*Dhv, (long long)Dhv);

  // 14) av /= den
  divide_kernel<<<rows/8, blk, 0, stream>>>(av, qp, kssum, rows, Ls, Hh, Mm, Dhv);

  // 15) out = av_flat @ Wo   (8192 x 1024 x 1024)
  wmma_gemm<false,false,128><<<g1, blk, 0, stream>>>(av, Wo, out, 1.0f,
                                                     Dm, Dm, Dm, Dm,
                                                     1, 0,0, 0,0, 0,0);
}
